// LSNNLayer_3882650436360
// MI455X (gfx1250) — matrix operand order repaired and checked by reading, compiled, not run on hardware
//
#include <hip/hip_runtime.h>
#include <hip/hip_bf16.h>

// ---------------------------------------------------------------------------
// LSNN persistent-scan kernel for gfx1250 (MI455X).
//   T=1000, B=64, N_in=256, N_h=512.
//   Pre-pass: convert input tensor to bf16 once (streaming, parallel).
//   Scan: 16 WGs x 256 threads (8 wave32 waves). WG g owns hidden cols
//   [32g,32g+32). Each wave owns one 16x16 C-tile. Weight B-fragments are
//   register-resident for the whole scan (192 VGPRs/wave); A-fragments are
//   bulk-preloaded (Ax[8], Az[8]) and the schedule is pinned with
//   sched_group_barrier so >=8 fragment loads stay in flight ahead of every
//   WMMA. v/i/b state lives in registers in the WMMA C layout. Spikes are
//   exchanged via a contiguous double-buffered bf16 buffer in d_ws +
//   agent-scope grid barrier.
// ---------------------------------------------------------------------------

typedef __attribute__((ext_vector_type(16))) __bf16          v16bf;
typedef __attribute__((ext_vector_type(16))) unsigned short  v16u16;
typedef __attribute__((ext_vector_type(8)))  unsigned int    v8u32;
typedef __attribute__((ext_vector_type(8)))  float           v8f;

#define T_STEPS 1000
#define BATCH   64
#define N_IN    256
#define N_H     512
#define NWG     16
#define HS      32                 // hidden slice per workgroup
#define ZELEMS  (BATCH * N_H)      // 32768 elements per z buffer

// sched_group_barrier masks
#define SGB_MATRIX    0x008
#define SGB_VMEM_READ 0x020

// round-to-nearest-even f32 -> bf16
__device__ __forceinline__ unsigned short f2bf(float f) {
  unsigned int u = __builtin_bit_cast(unsigned int, f);
  u += 0x7FFFu + ((u >> 16) & 1u);
  return (unsigned short)(u >> 16);
}

__device__ __forceinline__ v16bf pack2(uint4 a, uint4 b) {
  v8u32 t;
  t[0] = a.x; t[1] = a.y; t[2] = a.z; t[3] = a.w;
  t[4] = b.x; t[5] = b.y; t[6] = b.z; t[7] = b.w;
  return __builtin_bit_cast(v16bf, t);
}

// A-fragment from bf16 global: 16B chunk at p, 16B chunk at p+16 elements
__device__ __forceinline__ v16bf load_a_bf16(const unsigned short* p) {
  uint4 lo = *(const uint4*)(p);
  uint4 hi = *(const uint4*)(p + 16);
  return pack2(lo, hi);
}

// A-fragment from f32 global (fallback path): 8 floats at p, 8 at p+16
__device__ __forceinline__ v16bf load_a_f32(const float* p) {
  float4 f0 = ((const float4*)p)[0];
  float4 f1 = ((const float4*)p)[1];
  float4 f2 = ((const float4*)(p + 16))[0];
  float4 f3 = ((const float4*)(p + 16))[1];
  v16u16 u;
  u[0]  = f2bf(f0.x); u[1]  = f2bf(f0.y); u[2]  = f2bf(f0.z); u[3]  = f2bf(f0.w);
  u[4]  = f2bf(f1.x); u[5]  = f2bf(f1.y); u[6]  = f2bf(f1.z); u[7]  = f2bf(f1.w);
  u[8]  = f2bf(f2.x); u[9]  = f2bf(f2.y); u[10] = f2bf(f2.z); u[11] = f2bf(f2.w);
  u[12] = f2bf(f3.x); u[13] = f2bf(f3.y); u[14] = f2bf(f3.z); u[15] = f2bf(f3.w);
  return __builtin_bit_cast(v16bf, u);
}

// B-fragment from LDS: 32B contiguous per lane
__device__ __forceinline__ v16bf load_b_lds(const unsigned short* p) {
  uint4 lo = ((const uint4*)p)[0];
  uint4 hi = ((const uint4*)p)[1];
  return pack2(lo, hi);
}

__device__ __forceinline__ v8f wmma_bf16(v16bf a, v16bf b, v8f c) {
  // 8 args: (neg_a, A, neg_b, B, c_mod, C, reuse_a, reuse_b)
  return __builtin_amdgcn_wmma_f32_16x16x32_bf16(false, a, false, b,
                                                 (short)0, c, false, false);
}

// Agent-scope grid barrier: release add, acquire spin (global_inv on exit).
__device__ __forceinline__ void grid_barrier(unsigned int* ctr, unsigned int target) {
  __threadfence();
  __syncthreads();
  if (threadIdx.x == 0) {
    __hip_atomic_fetch_add(ctr, 1u, __ATOMIC_RELEASE, __HIP_MEMORY_SCOPE_AGENT);
    while (__hip_atomic_load(ctr, __ATOMIC_ACQUIRE, __HIP_MEMORY_SCOPE_AGENT) < target) {
      __builtin_amdgcn_s_sleep(1);
    }
  }
  __syncthreads();
  __builtin_amdgcn_fence(__ATOMIC_ACQUIRE, "agent");
}

__global__ void lsnn_init(unsigned int* ctr) {
  if (threadIdx.x == 0) *ctr = 0u;
}

// Streaming pre-pass: x (f32) -> xbf (bf16). One float4 -> packed uint2.
__global__ __launch_bounds__(256)
void lsnn_cvt_x(const float* __restrict__ x, unsigned short* __restrict__ xbf,
                int n4) {
  int i = blockIdx.x * blockDim.x + threadIdx.x;
  if (i < n4) {
    float4 f = ((const float4*)x)[i];
    uint2 o;
    o.x = (unsigned int)f2bf(f.x) | ((unsigned int)f2bf(f.y) << 16);
    o.y = (unsigned int)f2bf(f.z) | ((unsigned int)f2bf(f.w) << 16);
    ((uint2*)xbf)[i] = o;
  }
}

template <bool XBF>
__global__ __launch_bounds__(256, 1)
void lsnn_persistent(const void* __restrict__ xin,
                     const float* __restrict__ z0,
                     const float* __restrict__ v0,
                     const float* __restrict__ i0,
                     const float* __restrict__ b0,
                     const float* __restrict__ w_in,
                     const float* __restrict__ w_rec,
                     float* __restrict__ out,
                     unsigned int* __restrict__ ctr,
                     unsigned short* __restrict__ zbase) {
  // bf16 B-fragments: [col_tile][k_chunk][lane][16 bf16]
  __shared__ __align__(16) unsigned short s_win [2][ 8][32][16];  // 16 KB
  __shared__ __align__(16) unsigned short s_wrec[2][16][32][16];  // 32 KB

  const int tid  = threadIdx.x;
  const int lane = tid & 31;
  const int wv   = tid >> 5;   // 0..7
  const int r    = wv >> 1;    // batch row tile 0..3
  const int ct   = wv & 1;     // hidden col tile 0..1
  const int H0   = blockIdx.x * HS;

  // ---- stage w_in slice as B-fragments (B layout: lane<16 -> K{0..7,16..23} of
  // column lane; lane>=16 -> K{8..15,24..31} of column lane-16) ----
  for (int e = tid; e < 2 * 8 * 32; e += 256) {
    int ect = e >> 8, rem = e & 255;
    int ekc = rem >> 5, eln = rem & 31;
    int col = H0 + ect * 16 + (eln & 15);
    int kb  = ekc * 32 + ((eln >= 16) ? 8 : 0);
    const float* src = w_in + col * N_IN + kb;
    #pragma unroll
    for (int j = 0; j < 16; ++j) s_win[ect][ekc][eln][j] = f2bf(src[j < 8 ? j : j + 8]);
  }
  // ---- stage w_rec slice ----
  for (int e = tid; e < 2 * 16 * 32; e += 256) {
    int ect = e >> 9, rem = e & 511;
    int ekc = rem >> 5, eln = rem & 31;
    int col = H0 + ect * 16 + (eln & 15);
    int kb  = ekc * 32 + ((eln >= 16) ? 8 : 0);
    const float* src = w_rec + col * N_H + kb;
    #pragma unroll
    for (int j = 0; j < 16; ++j) s_wrec[ect][ekc][eln][j] = f2bf(src[j < 8 ? j : j + 8]);
  }

  // ---- per-lane state mapping (WMMA 16x16 f32 C layout) ----
  const int mOff    = (lane >> 4) << 3;          // 0 or 8
  const int nCol    = H0 + ct * 16 + (lane & 15);
  const int rowBase = r * 16 + mOff;

  v8f vS, iS, bS;
  #pragma unroll
  for (int j = 0; j < 8; ++j) {
    int idx = (rowBase + j) * N_H + nCol;
    vS[j] = v0[idx];
    iS[j] = i0[idx];
    bS[j] = b0[idx];
    zbase[idx] = f2bf(z0[idx]);   // seed z double-buffer (buffer 0) with z0
  }

  // ---- A-fragment addressing (A layout: lane<16 -> K{0..7,16..23}) ----
  const int aRow = r * 16 + (lane & 15);
  const int aK   = (lane >> 4) * 8;   // 0 or 8
  const int aZ   = aRow * N_H + aK;   // per-lane offset within a z buffer

  unsigned int phase = 1;
  grid_barrier(ctr, NWG * phase); ++phase;   // z0 + states + LDS ready

  // ---- hoist weight B-fragments into registers for the whole scan ----
  v16bf Bwin[8], Bwrec[16];
  #pragma unroll
  for (int kc = 0; kc < 8; ++kc)  Bwin[kc]  = load_b_lds(&s_win [ct][kc][lane][0]);
  #pragma unroll
  for (int kc = 0; kc < 16; ++kc) Bwrec[kc] = load_b_lds(&s_wrec[ct][kc][lane][0]);

  const unsigned short* xb = (const unsigned short*)xin;  // XBF path
  const float*          xf = (const float*)xin;           // fallback path

  v8f zN = {};
  for (int t = 0; t < T_STEPS; ++t) {
    // contiguous double buffer: read (t&1), write (t+1)&1
    const unsigned short* zrow  = zbase + (t & 1) * ZELEMS + aZ;
    unsigned short*       znext = zbase + ((t + 1) & 1) * ZELEMS;
    const size_t xoff = ((size_t)t * BATCH + aRow) * N_IN + aK;

    // ---------------- GEMM: x_t @ w_in.T  +  z_prev @ w_rec.T --------------
    // Bulk-preload Ax (all 8 input chunks) + Az (first 8 recurrent chunks);
    // trailing 8 recurrent chunks stream in during the z-WMMA loop.
    v16bf Ax[8];
    #pragma unroll
    for (int i = 0; i < 8; ++i)
      Ax[i] = XBF ? load_a_bf16(xb + xoff + i * 32)
                  : load_a_f32 (xf + xoff + i * 32);
    v16bf Az[8];
    #pragma unroll
    for (int i = 0; i < 8; ++i)
      Az[i] = load_a_bf16(zrow + i * 32);

    v8f acc = {};
    #pragma unroll
    for (int i = 0; i < 8; ++i)
      acc = wmma_bf16(Ax[i], Bwin[i], acc);
    #pragma unroll
    for (int i = 0; i < 16; ++i) {
      acc = wmma_bf16(Az[i & 7], Bwrec[i], acc);
      if (i < 8) Az[i] = load_a_bf16(zrow + (i + 8) * 32);
    }

    // Pin the pipeline shape: 32 fragment loads issued up front, then the
    // 24 WMMAs with the trailing 16 z-loads interleaved 2-per-WMMA through
    // the middle slots (>=8 fragments always in flight ahead of each WMMA).
    __builtin_amdgcn_sched_group_barrier(SGB_VMEM_READ, 32, 0);
    #pragma unroll
    for (int i = 0; i < 8; ++i)
      __builtin_amdgcn_sched_group_barrier(SGB_MATRIX, 1, 0);
    #pragma unroll
    for (int i = 0; i < 8; ++i) {
      __builtin_amdgcn_sched_group_barrier(SGB_MATRIX, 1, 0);
      __builtin_amdgcn_sched_group_barrier(SGB_VMEM_READ, 2, 0);
    }
    #pragma unroll
    for (int i = 0; i < 8; ++i)
      __builtin_amdgcn_sched_group_barrier(SGB_MATRIX, 1, 0);
    // Fence the region so prefetches / epilogue stores can't enter the
    // pattern above.
    __builtin_amdgcn_sched_barrier(0);

    // prefetch next step's input rows (global_prefetch_b8) into L2
    if (t + 1 < T_STEPS) {
      const size_t xnoff = ((size_t)(t + 1) * BATCH + aRow) * N_IN + aK;
      #pragma unroll
      for (int kc = 0; kc < 8; ++kc) {
        if (XBF) __builtin_prefetch(xb + xnoff + kc * 32, 0, 3);
        else     __builtin_prefetch(xf + xnoff + kc * 32, 0, 3);
      }
    }

    // ---------------- LSNN state update (fp32, in registers) ---------------
    // DT*TAU_MEM_INV=0.1  DT*TAU_SYN_INV=0.2  DT*TAU_ADAPT_INV=1.25e-6
    // TAU_ADAPT_INV*BETA = 1.8/800 = 0.00225
    v8f v_dec = vS + 0.1f * (iS - vS);
    v8f i_dec = iS * 0.8f;
    v8f b_dec = bS + 1.25e-6f * (1.0f - bS);
    #pragma unroll
    for (int j = 0; j < 8; ++j)
      zN[j] = (v_dec[j] - b_dec[j]) > 0.0f ? 1.0f : 0.0f;
    vS = (1.0f - zN) * v_dec;       // V_RESET = 0
    iS = i_dec + acc;
    bS = b_dec + zN * 0.00225f;

    // ---------------- publish spikes -----------------
    float* orow = out + (size_t)t * (BATCH * N_H);
    #pragma unroll
    for (int j = 0; j < 8; ++j) {
      int idx = (rowBase + j) * N_H + nCol;
      orow[idx]  = zN[j];
      znext[idx] = f2bf(zN[j]);
    }

    grid_barrier(ctr, NWG * phase); ++phase;
  }

  // ---- final states: d_out = [outputs | zT | vT | iT | bT] ----
  const size_t o0 = (size_t)T_STEPS * BATCH * N_H;
  #pragma unroll
  for (int j = 0; j < 8; ++j) {
    int idx = (rowBase + j) * N_H + nCol;
    out[o0 + 0 * (BATCH * N_H) + idx] = zN[j];
    out[o0 + 1 * (BATCH * N_H) + idx] = vS[j];
    out[o0 + 2 * (BATCH * N_H) + idx] = iS[j];
    out[o0 + 3 * (BATCH * N_H) + idx] = bS[j];
  }
}

extern "C" void kernel_launch(void* const* d_in, const int* in_sizes, int n_in,
                              void* d_out, int out_size, void* d_ws, size_t ws_size,
                              hipStream_t stream) {
  (void)in_sizes; (void)n_in; (void)out_size;
  const float* x     = (const float*)d_in[0];
  const float* z0    = (const float*)d_in[1];
  const float* v0    = (const float*)d_in[2];
  const float* i0    = (const float*)d_in[3];
  const float* b0    = (const float*)d_in[4];
  const float* w_in  = (const float*)d_in[5];
  const float* w_rec = (const float*)d_in[6];
  float* out = (float*)d_out;

  // d_ws layout: [counter (256B) | zbuf x2 contiguous 128KB | xbf 31.25MiB]
  unsigned char* ws = (unsigned char*)d_ws;
  unsigned int*   ctr   = (unsigned int*)ws;
  unsigned short* zbase = (unsigned short*)(ws + 256);
  unsigned short* xbf   = (unsigned short*)(ws + 256 + 2 * (size_t)ZELEMS * 2);

  const size_t xElems  = (size_t)T_STEPS * BATCH * N_IN;         // 16.384M
  const size_t needBf  = 256 + 2 * (size_t)ZELEMS * 2 + xElems * 2;

  lsnn_init<<<1, 64, 0, stream>>>(ctr);

  if (ws_size >= needBf) {
    const int n4 = (int)(xElems / 4);                            // 4.096M
    lsnn_cvt_x<<<(n4 + 255) / 256, 256, 0, stream>>>(x, xbf, n4);
    lsnn_persistent<true><<<NWG, 256, 0, stream>>>(
        (const void*)xbf, z0, v0, i0, b0, w_in, w_rec, out, ctr, zbase);
  } else {
    lsnn_persistent<false><<<NWG, 256, 0, stream>>>(
        (const void*)x, z0, v0, i0, b0, w_in, w_rec, out, ctr, zbase);
  }
}
